// RNNLanguageModel_89464168775747
// MI455X (gfx1250) — matrix operand order repaired and checked by reading, not compiled, not run on hardware
//
#include <hip/hip_runtime.h>
#include <hip/hip_bf16.h>

// ---------------------------------------------------------------------------
// RNN LM on gfx1250: bf16 WMMA (v_wmma_f32_16x16x32_bf16) for all GEMMs,
// single-workgroup LDS-resident recurrent scan.
// Shapes: B=16, T=512, H=256, V=32000.  M-dim index m = b*T + t.
// ---------------------------------------------------------------------------

typedef __attribute__((ext_vector_type(16))) __bf16 v16bf;
typedef __attribute__((ext_vector_type(8)))  float  v8f;

#define Bn   16
#define Tn   512
#define Hn   256
#define Vn   32000
#define Mn   (Bn * Tn)      // 8192 rows
#define HPAD 264            // LDS row pad: 528B stride -> conflict-free lanes

// float -> bf16 round-to-nearest-even
__device__ __forceinline__ unsigned short f2bf(float f) {
    unsigned int u = __builtin_bit_cast(unsigned int, f);
    u = (u + 0x7FFFu + ((u >> 16) & 1u)) >> 16;
    return (unsigned short)u;
}

union FragU { uint4 q[2]; v16bf v; };

// A-matrix fragment (16x32 bf16), src row-major [*, ld].
// Lane l: M = m0 + (l&15); K = k0 + {0..7,16..23} (l<16) or {8..15,24..31}.
__device__ __forceinline__ v16bf load_a_frag(const unsigned short* base, int ld,
                                             int m0, int k0) {
    int l   = threadIdx.x & 31;
    int m   = m0 + (l & 15);
    int klo = k0 + ((l >> 4) << 3);            // +0 or +8
    const unsigned short* p = base + (size_t)m * ld + klo;
    FragU u;
    u.q[0] = *reinterpret_cast<const uint4*>(p);        // K klo..klo+7
    u.q[1] = *reinterpret_cast<const uint4*>(p + 16);   // K klo+16..klo+23
    return u.v;
}

// B-matrix fragment (32x16 bf16) from row-major weights W[n][k] (B[k][n]=W[n][k]).
// Lane l: N = n0 + (l&15); K = k0 + {0..7,16..23} (l<16) or {8..15,24..31}.
__device__ __forceinline__ v16bf load_b_frag(const unsigned short* base, int ld,
                                             int n0, int k0) {
    int l  = threadIdx.x & 31;
    int n  = n0 + (l & 15);
    int kb = k0 + ((l >> 4) << 3);             // +0 or +8
    const unsigned short* p = base + (size_t)n * ld + kb;
    FragU u;
    u.q[0] = *reinterpret_cast<const uint4*>(p);
    u.q[1] = *reinterpret_cast<const uint4*>(p + 16);
    return u.v;
}

// ---------------------------------------------------------------------------
// Kernel 1: embedding gather + cast to bf16.  dst[m][h] = bf16(embed[x[m]][h])
__global__ void __launch_bounds__(256) embed_cast_kernel(
    const int* __restrict__ x, const float* __restrict__ embed,
    unsigned short* __restrict__ dst) {
    int i   = blockIdx.x * 256 + threadIdx.x;   // 0 .. Mn*Hn-1
    int row = i >> 8;
    int h   = i & 255;
    int tok = x[row];
    dst[i] = f2bf(embed[(size_t)tok * Hn + h]);
}

// Kernel 2: generic fp32 -> bf16 cast (weights)
__global__ void __launch_bounds__(256) cast_kernel(
    const float* __restrict__ src, unsigned short* __restrict__ dst, int n) {
    int i = blockIdx.x * 256 + threadIdx.x;
    if (i < n) dst[i] = f2bf(src[i]);
}

// ---------------------------------------------------------------------------
// Kernel 3: pre[m][n] = sum_h emb_bf[m][h]*W_ih_bf[n][h] + b_ih[n] + b_hh[n]
// One wave per 16x16 tile; 8192 waves total.
__global__ void __launch_bounds__(256) pre_gemm_kernel(
    const unsigned short* __restrict__ A, const unsigned short* __restrict__ W,
    const float* __restrict__ b_ih, const float* __restrict__ b_hh,
    float* __restrict__ C) {
    int wave = (blockIdx.x * 256 + threadIdx.x) >> 5;   // 0..8191
    int m0 = (wave >> 4) << 4;                          // 512 M-tiles
    int n0 = (wave & 15) << 4;                          // 16  N-tiles
    int l  = threadIdx.x & 31;
    int n  = n0 + (l & 15);
    float bias = b_ih[n] + b_hh[n];
    v8f c;
#pragma unroll
    for (int r = 0; r < 8; ++r) c[r] = bias;
#pragma unroll
    for (int k0 = 0; k0 < Hn; k0 += 32) {
        v16bf a = load_a_frag(A, Hn, m0, k0);
        v16bf b = load_b_frag(W, Hn, n0, k0);
        c = __builtin_amdgcn_wmma_f32_16x16x32_bf16(false, a, false, b,
                                                    (short)0, c, false, false);
    }
    int mrow = m0 + ((l >> 4) << 3);
#pragma unroll
    for (int r = 0; r < 8; ++r)
        C[(size_t)(mrow + r) * Hn + n] = c[r];
}

// ---------------------------------------------------------------------------
// Kernel 4: sequential scan  h_t = tanh(pre_t + h_{t-1} @ W_hh^T)
// Single workgroup, 16 waves; wave w owns output columns [16w, 16w+16).
// h double-buffered in LDS (bf16, padded rows); W_hh_bf served from WGP$/L2.
// Emits hs[m][h] in bf16 for the logits GEMM.
__global__ void __launch_bounds__(512) rnn_scan_kernel(
    const float* __restrict__ pre, const unsigned short* __restrict__ Whh,
    unsigned short* __restrict__ hs) {
    __shared__ unsigned short hbuf[2][Bn][HPAD];
    int tid   = threadIdx.x;
    int wave  = tid >> 5;
    int l     = tid & 31;
    int n0    = wave << 4;
    int n     = n0 + (l & 15);
    int bbase = (l >> 4) << 3;                 // 0 or 8 (C-layout row base)

    for (int i = tid; i < 2 * Bn * HPAD; i += 512)
        (&hbuf[0][0][0])[i] = 0;               // h0 = 0
    __syncthreads();

    int cur = 0;
    for (int t = 0; t < Tn; ++t) {
        v8f c;
#pragma unroll
        for (int r = 0; r < 8; ++r) {
            int b = bbase + r;
            c[r] = pre[((size_t)b * Tn + t) * Hn + n];
        }
#pragma unroll
        for (int k0 = 0; k0 < Hn; k0 += 32) {
            v16bf a = load_a_frag(&hbuf[cur][0][0], HPAD, 0, k0);
            v16bf b = load_b_frag(Whh, Hn, n0, k0);
            c = __builtin_amdgcn_wmma_f32_16x16x32_bf16(false, a, false, b,
                                                        (short)0, c, false, false);
        }
        int nxt = cur ^ 1;
#pragma unroll
        for (int r = 0; r < 8; ++r) {
            int b = bbase + r;
            unsigned short hb = f2bf(tanhf(c[r]));
            hbuf[nxt][b][n0 + (l & 15)] = hb;            // feed next step
            hs[((size_t)b * Tn + t) * Hn + n] = hb;      // activations for FC
        }
        __syncthreads();
        cur = nxt;
    }
}

// ---------------------------------------------------------------------------
// Kernel 5: logits[m][v] = sum_h hs_bf[m][h]*W_fc_bf[v][h] + b_fc[v]
// Each wave computes a 32x64 register tile (2 M-tiles x 4 N-tiles):
// 8 WMMAs per K-chunk for 12 b128 loads -> 1.5 loads/WMMA from L2.
__global__ void __launch_bounds__(256) logits_kernel(
    const unsigned short* __restrict__ Hs, const unsigned short* __restrict__ Wfc,
    const float* __restrict__ bfc, float* __restrict__ out) {
    int gw = (blockIdx.x * 256 + threadIdx.x) >> 5;     // 0..127999
    const int nstrips = Vn / 64;                        // 500
    int m0 = (gw / nstrips) << 5;                       // 256 M-blocks of 32
    int n0 = (gw % nstrips) << 6;                       // 500 N-strips of 64
    int l  = threadIdx.x & 31;
    v8f c[2][4];
#pragma unroll
    for (int j = 0; j < 4; ++j) {
        float bias = bfc[n0 + j * 16 + (l & 15)];
#pragma unroll
        for (int r = 0; r < 8; ++r) { c[0][j][r] = bias; c[1][j][r] = bias; }
    }
    // lane's B row base (for prefetching the next K-chunk into cache)
    const unsigned short* brow = Wfc + (size_t)(n0 + (l & 15)) * Hn;
    for (int k0 = 0; k0 < Hn; k0 += 32) {
        if (k0 + 32 < Hn)
            __builtin_prefetch(brow + k0 + 32, 0, 0);   // global_prefetch_b8
        v16bf a0 = load_a_frag(Hs, Hn, m0,      k0);
        v16bf a1 = load_a_frag(Hs, Hn, m0 + 16, k0);
#pragma unroll
        for (int j = 0; j < 4; ++j) {
            v16bf b = load_b_frag(Wfc, Hn, n0 + j * 16, k0);
            c[0][j] = __builtin_amdgcn_wmma_f32_16x16x32_bf16(false, a0, false, b,
                                                              (short)0, c[0][j], false, false);
            c[1][j] = __builtin_amdgcn_wmma_f32_16x16x32_bf16(false, a1, false, b,
                                                              (short)0, c[1][j], false, false);
        }
    }
#pragma unroll
    for (int i = 0; i < 2; ++i) {
        int mrow = m0 + i * 16 + ((l >> 4) << 3);
#pragma unroll
        for (int j = 0; j < 4; ++j)
#pragma unroll
            for (int r = 0; r < 8; ++r)
                out[(size_t)(mrow + r) * Vn + n0 + j * 16 + (l & 15)] = c[i][j][r];
    }
}

// ---------------------------------------------------------------------------
extern "C" void kernel_launch(void* const* d_in, const int* in_sizes, int n_in,
                              void* d_out, int out_size, void* d_ws, size_t ws_size,
                              hipStream_t stream) {
    const int*   x     = (const int*)  d_in[0];
    const float* embed = (const float*)d_in[1];
    const float* W_ih  = (const float*)d_in[2];
    const float* W_hh  = (const float*)d_in[3];
    const float* b_ih  = (const float*)d_in[4];
    const float* b_hh  = (const float*)d_in[5];
    const float* W_fc  = (const float*)d_in[6];
    const float* b_fc  = (const float*)d_in[7];
    float* out = (float*)d_out;

    // Workspace layout (bytes), all offsets 256B-aligned:
    char* ws = (char*)d_ws;
    unsigned short* emb_bf = (unsigned short*)(ws + 0);             //  4,194,304
    unsigned short* wih_bf = (unsigned short*)(ws + 4194304);       //    131,072
    unsigned short* whh_bf = (unsigned short*)(ws + 4325376);       //    131,072
    unsigned short* wfc_bf = (unsigned short*)(ws + 4456448);       // 16,384,000
    float*          pre    = (float*)         (ws + 20840448);      //  8,388,608
    unsigned short* hs_bf  = (unsigned short*)(ws + 29229056);      //  4,194,304
    // total ~33.4 MB

    embed_cast_kernel<<<(Mn * Hn) / 256, 256, 0, stream>>>(x, embed, emb_bf);
    cast_kernel<<<(Hn * Hn) / 256, 256, 0, stream>>>(W_ih, wih_bf, Hn * Hn);
    cast_kernel<<<(Hn * Hn) / 256, 256, 0, stream>>>(W_hh, whh_bf, Hn * Hn);
    cast_kernel<<<(Vn * Hn) / 256, 256, 0, stream>>>(W_fc, wfc_bf, Vn * Hn);

    pre_gemm_kernel<<<(Mn / 16) * (Hn / 16) / 8, 256, 0, stream>>>(
        emb_bf, wih_bf, b_ih, b_hh, pre);

    rnn_scan_kernel<<<1, 512, 0, stream>>>(pre, whh_bf, hs_bf);

    logits_kernel<<<(Mn / 32) * (Vn / 64) / 8, 256, 0, stream>>>(
        hs_bf, wfc_bf, b_fc, out);
}
